// ASPPGraphFusion_16381005267601
// MI455X (gfx1250) — matrix operand order repaired and checked by reading, not compiled, not run on hardware
//
#include <hip/hip_runtime.h>

typedef __attribute__((ext_vector_type(16))) _Float16 v16h;
typedef __attribute__((ext_vector_type(8)))  _Float16 v8h;
typedef __attribute__((ext_vector_type(8)))  float    v8f;

#define B_   8
#define CIN  32
#define CMID 16
#define COUT 32
#define H_   384
#define W_   384
#define HW_  (H_*W_)
#define NTAP 25

// workspace byte offsets (all 256B aligned); total required = 517632 B
#define WS_STATS 0          // 256*13 f32
#define WS_S     13312      // 256*25 f32
#define WS_WM    38912      // 16*32*25 f32
#define WS_G     90112      // 8*16 f32
#define WS_C     90624      // 8*32 f32
#define WS_FG    91648      // 8*32*16 f32
#define WS_WBA   108032     // 8*25*2*32*16 f16 = 409600 B

#define WSCALE 256.0f       // keep folded weights out of f16 subnormal range

__device__ __forceinline__ void tap_offsets(int t, int& dy, int& dx) {
  const int OY8[8] = {-1,-1,-1, 0, 0, 1, 1, 1};
  const int OX8[8] = {-1, 0, 1,-1, 1,-1, 0, 1};
  if (t == 0) { dy = 0; dx = 0; }
  else { int d = (t-1)/8 + 1; int r8 = (t-1)&7; dy = OY8[r8]*d; dx = OX8[r8]*d; }
}

__device__ __forceinline__ v16h ld16h(const _Float16* p) {
  v8h lo = *(const v8h*)p;
  v8h hi = *(const v8h*)(p + 8);
  return __builtin_shufflevector(lo, hi, 0,1,2,3,4,5,6,7,8,9,10,11,12,13,14,15);
}

// ---------------------------------------------------------------------------
// K1: per (b,c) plane stats: total, 3 top/bottom row sums, 3 left/right col sums
// ---------------------------------------------------------------------------
__global__ void __launch_bounds__(256) k1_stats(const float* __restrict__ x,
                                                float* __restrict__ stats) {
  int bc = blockIdx.x;                               // 0..255 = b*32+c
  const float4* p = (const float4*)(x + (size_t)bc * HW_);
  float tot=0, rt0=0, rt1=0, rt2=0, rb0=0, rb1=0, rb2=0;
  float cl0=0, cl1=0, cl2=0, cr0=0, cr1=0, cr2=0;
  for (int q = threadIdx.x; q < HW_/4; q += 256) {
    float4 v = p[q];
    int r  = q / (W_/4);
    int c4 = (q - r*(W_/4)) * 4;
    float s4 = v.x + v.y + v.z + v.w;
    tot += s4;
    if (r == 0) rt0 += s4; else if (r == 1) rt1 += s4; else if (r == 2) rt2 += s4;
    if (r == H_-3) rb0 += s4; else if (r == H_-2) rb1 += s4; else if (r == H_-1) rb2 += s4;
    if (c4 == 0)        { cl0 += v.x; cl1 += v.y; cl2 += v.z; }
    else if (c4 == W_-4){ cr0 += v.y; cr1 += v.z; cr2 += v.w; }
  }
  __shared__ float sacc[13];
  if (threadIdx.x < 13) sacc[threadIdx.x] = 0.f;
  __syncthreads();
  atomicAdd(&sacc[0],  tot);
  atomicAdd(&sacc[1],  rt0); atomicAdd(&sacc[2],  rt1); atomicAdd(&sacc[3],  rt2);
  atomicAdd(&sacc[4],  rb0); atomicAdd(&sacc[5],  rb1); atomicAdd(&sacc[6],  rb2);
  atomicAdd(&sacc[7],  cl0); atomicAdd(&sacc[8],  cl1); atomicAdd(&sacc[9],  cl2);
  atomicAdd(&sacc[10], cr0); atomicAdd(&sacc[11], cr1); atomicAdd(&sacc[12], cr2);
  __syncthreads();
  if (threadIdx.x < 13) stats[bc*13 + threadIdx.x] = sacc[threadIdx.x];
}

// ---------------------------------------------------------------------------
// K2a: shifted-sum S[b,c,tap] via inclusion-exclusion over zero padding
// ---------------------------------------------------------------------------
__device__ __forceinline__ void prefix2(float P[3][3]) {
  for (int i = 1; i < 3; ++i)
    for (int j = 0; j < 3; ++j) P[i][j] += P[i-1][j];
  for (int i = 0; i < 3; ++i)
    for (int j = 1; j < 3; ++j) P[i][j] += P[i][j-1];
}

__global__ void __launch_bounds__(256) k2a_shift(const float* __restrict__ x,
                                                 const float* __restrict__ stats,
                                                 float* __restrict__ S) {
  int t = threadIdx.x;                               // (b,c) pair
  const float* st = stats + t*13;
  float tot = st[0];
  float rt[3] = {st[1], st[2], st[3]};
  float rb[3] = {st[4], st[5], st[6]};
  float cl[3] = {st[7], st[8], st[9]};
  float cr[3] = {st[10], st[11], st[12]};
  const float* px = x + (size_t)t * HW_;
  float Ptl[3][3], Ptr[3][3], Pbl[3][3], Pbr[3][3];
  for (int i = 0; i < 3; ++i)
    for (int j = 0; j < 3; ++j) {
      Ptl[i][j] = px[i*W_ + j];
      Ptr[i][j] = px[i*W_ + (W_-1-j)];
      Pbl[i][j] = px[(H_-1-i)*W_ + j];
      Pbr[i][j] = px[(H_-1-i)*W_ + (W_-1-j)];
    }
  prefix2(Ptl); prefix2(Ptr); prefix2(Pbl); prefix2(Pbr);
#pragma unroll
  for (int tp = 0; tp < NTAP; ++tp) {
    int dy, dx; tap_offsets(tp, dy, dx);
    float rband = 0.f, cband = 0.f, corner = 0.f;
    if (dy > 0) { for (int k = 0; k <  dy; ++k) rband += rt[k]; }
    if (dy < 0) { for (int k = 0; k < -dy; ++k) rband += rb[2-k]; }
    if (dx > 0) { for (int k = 0; k <  dx; ++k) cband += cl[k]; }
    if (dx < 0) { for (int k = 0; k < -dx; ++k) cband += cr[2-k]; }
    if (dy > 0 && dx > 0) corner = Ptl[dy-1][dx-1];
    if (dy > 0 && dx < 0) corner = Ptr[dy-1][-dx-1];
    if (dy < 0 && dx > 0) corner = Pbl[-dy-1][dx-1];
    if (dy < 0 && dx < 0) corner = Pbr[-dy-1][-dx-1];
    S[t*NTAP + tp] = tot - rband - cband + corner;
  }
}

// ---------------------------------------------------------------------------
// K2b: node means -> GCN g -> softmax -> merged tap kernel Wm, constants C, Fg
// ---------------------------------------------------------------------------
__global__ void __launch_bounds__(256) k2b_fold(
    const float* __restrict__ S,
    const float* __restrict__ w1, const float* __restrict__ b1,
    const float* __restrict__ w2, const float* __restrict__ b2,
    const float* __restrict__ w3, const float* __restrict__ b3,
    const float* __restrict__ w4, const float* __restrict__ b4,
    const float* __restrict__ w5, const float* __restrict__ b5,
    const float* __restrict__ gcn_w, const float* __restrict__ gcn_b,
    const float* __restrict__ attn,
    const float* __restrict__ fw, const float* __restrict__ fb,
    float* __restrict__ wsWm, float* __restrict__ wsG,
    float* __restrict__ wsC, float* __restrict__ wsFg) {
  __shared__ float sNode[5][128];
  __shared__ float sNbar[128];
  __shared__ float sG[128];
  __shared__ float sK[128];
  __shared__ float sP[5];
  const int RIDX[3][3] = {{0,1,2},{3,-1,4},{5,6,7}};
  int t = threadIdx.x;
  if (t == 0) {                                      // softmax(attn)
    float m = attn[0];
    for (int i = 1; i < 5; ++i) m = fmaxf(m, attn[i]);
    float s = 0.f, e[5];
    for (int i = 0; i < 5; ++i) { e[i] = __expf(attn[i] - m); s += e[i]; }
    for (int i = 0; i < 5; ++i) sP[i] = e[i] / s;
  }
  if (t < 128) {                                     // node means per branch
    int b = t >> 4, c = t & 15;
    const float* Sb = S + (size_t)b * CIN * NTAP;
    float a = 0.f, a5 = 0.f;
    for (int cp = 0; cp < CIN; ++cp) {
      a  += w1[c*CIN + cp] * Sb[cp*NTAP];
      a5 += w5[c*CIN + cp] * Sb[cp*NTAP];
    }
    sNode[0][t] = a  * (1.f/HW_) + b1[c];
    sNode[4][t] = a5 * (1.f/HW_) + b5[c];
    for (int d = 1; d <= 3; ++d) {
      const float* wd = (d==1) ? w2 : (d==2) ? w3 : w4;
      const float* bd = (d==1) ? b2 : (d==2) ? b3 : b4;
      float s = 0.f;
      for (int cp = 0; cp < CIN; ++cp) {
        const float* wrow = wd + (c*CIN + cp)*9;
        const float* Srow = Sb + cp*NTAP;
        for (int ky = 0; ky < 3; ++ky)
          for (int kx = 0; kx < 3; ++kx) {
            int tap = (ky==1 && kx==1) ? 0 : ((d-1)*8 + 1 + RIDX[ky][kx]);
            s += wrow[ky*3+kx] * Srow[tap];
          }
      }
      sNode[d][t] = s * (1.f/HW_) + bd[c];
    }
  }
  __syncthreads();
  if (t < 128) {
    float nb = 0.f;
    for (int i = 0; i < 5; ++i) nb += sNode[i][t];
    sNbar[t] = nb * 0.2f;
  }
  __syncthreads();
  if (t < 128) {                                     // g = nbar@gcn_w + gcn_b ; K
    int b = t >> 4, c = t & 15;
    float g = gcn_b[c];
    for (int k = 0; k < CMID; ++k) g += sNbar[b*16 + k] * gcn_w[k*CMID + c];
    sG[t] = g;
    wsG[t] = g;
    sK[t] = sP[0]*b1[c] + sP[1]*b2[c] + sP[2]*b3[c] + sP[3]*b4[c]
          + sP[4]*sNode[4][t];
  }
  __syncthreads();
  // merged tap weights Wm[c][c'][tap] (softmax folded in)
  for (int e = t; e < CMID*CIN*NTAP; e += 256) {
    int c   = e / (CIN*NTAP);
    int rem = e - c*(CIN*NTAP);
    int cp  = rem / NTAP;
    int tap = rem - cp*NTAP;
    float v;
    if (tap == 0) {
      v = sP[0]*w1[c*CIN + cp] + sP[1]*w2[(c*CIN+cp)*9 + 4]
        + sP[2]*w3[(c*CIN+cp)*9 + 4] + sP[3]*w4[(c*CIN+cp)*9 + 4];
    } else {
      int d  = (tap-1)/8 + 1;
      int r8 = (tap-1)&7;
      const int KY[8] = {0,0,0,1,1,2,2,2};
      const int KX[8] = {0,1,2,0,2,0,1,2};
      const float* wd = (d==1) ? w2 : (d==2) ? w3 : w4;
      v = sP[d] * wd[(c*CIN+cp)*9 + KY[r8]*3 + KX[r8]];
    }
    wsWm[e] = v;
  }
  // C[b][o] and Fg[b][o][c]
  {
    int b = t >> 5, o = t & 31;
    float Cv = fb[o];
    for (int c = 0; c < CMID; ++c) {
      float f = fw[o*CMID + c] * sG[b*16 + c];
      Cv += f * sK[b*16 + c];
      wsFg[(b*COUT + o)*CMID + c] = f;
    }
    wsC[t] = Cv;
  }
}

// ---------------------------------------------------------------------------
// K2c: swizzle per-batch merged weights into WMMA A-fragment lane layout (f16)
// A 16x32 f16 layout as used here (contiguous k, B matches): lane L (M=L&15): K = (L>=16?16:0) + e
// ---------------------------------------------------------------------------
__global__ void __launch_bounds__(256) k2c_frag(const float* __restrict__ Fg,
                                                const float* __restrict__ Wm,
                                                _Float16* __restrict__ WbA) {
  int gid = blockIdx.x * 256 + threadIdx.x;          // exactly 204800 threads
  int el   = gid & 15;
  int lane = (gid >> 4) & 31;
  int m    = (gid >> 9) & 1;
  int bt   = gid >> 10;
  int t    = bt % 25;
  int b    = bt / 25;
  int o  = m*16 + (lane & 15);
  int cp = ((lane & 16) ? 16 : 0) + el;
  float v = 0.f;
  for (int c = 0; c < CMID; ++c)
    v += Fg[(b*COUT + o)*CMID + c] * Wm[(c*CIN + cp)*NTAP + t];
  WbA[gid] = (_Float16)(v * WSCALE);
}

// ---------------------------------------------------------------------------
// K3: implicit-GEMM 25-tap conv. Tile 12 rows x 32 cols, 8 waves x 3 groups.
// B fragment: LDS [r][col][c] f16, stride 40 halves (16B aligned, no bank conf)
// ---------------------------------------------------------------------------
#define TH 12
#define TW 32
#define LR 18      // TH+6
#define LC 38      // TW+6
#define LS 40      // c-stride in halves

__global__ void __launch_bounds__(256) k3_conv(const float* __restrict__ x,
                                               const _Float16* __restrict__ WbA,
                                               const float* __restrict__ Cc,
                                               float* __restrict__ out) {
  __shared__ _Float16 smem[LR*LC*LS];                // 54720 B
  int b = blockIdx.z, ty = blockIdx.y, tx = blockIdx.x;
  int h0 = ty*TH - 3, w0 = tx*TW - 3;
  int tid = threadIdx.x;

  for (int idx = tid; idx < LR*LC*CIN; idx += 256) {
    int c   = idx / (LR*LC);
    int rem = idx - c*(LR*LC);
    int r   = rem / LC, col = rem - r*LC;
    int h = h0 + r, w = w0 + col;
    float v = 0.f;
    if ((unsigned)h < (unsigned)H_ && (unsigned)w < (unsigned)W_)
      v = x[((size_t)(b*CIN + c)*H_ + h)*W_ + w];
    smem[(r*LC + col)*LS + c] = (_Float16)v;
  }
  __syncthreads();

  int wave  = tid >> 5, lane = tid & 31;
  int laneN = lane & 15;
  int koff  = (lane & 16) ? 16 : 0;                  // B-fragment K offset
  int rowj[3], cgj[3];
#pragma unroll
  for (int j = 0; j < 3; ++j) {
    int gi = wave*3 + j;
    rowj[j] = gi >> 1;
    cgj[j]  = gi & 1;
  }
  v8f acc[3][2];
#pragma unroll
  for (int j = 0; j < 3; ++j)
    for (int m = 0; m < 2; ++m)
      for (int r = 0; r < 8; ++r) acc[j][m][r] = 0.f;

  const _Float16* Ab = WbA + (size_t)b * NTAP * 2 * 32 * 16;
#pragma unroll
  for (int t = 0; t < NTAP; ++t) {
    int dy, dx; tap_offsets(t, dy, dx);
    const _Float16* ap = Ab + ((t*2 + 0)*32 + lane)*16;
    v16h a0 = ld16h(ap);
    v16h a1 = ld16h(ap + 32*16);
#pragma unroll
    for (int j = 0; j < 3; ++j) {
      int r_in = rowj[j] + 3 + dy;
      int c_in = cgj[j]*16 + laneN + 3 + dx;
      v16h bf = ld16h(&smem[(r_in*LC + c_in)*LS + koff]);
      acc[j][0] = __builtin_amdgcn_wmma_f32_16x16x32_f16(
          false, a0, false, bf, (short)0, acc[j][0], false, false);
      acc[j][1] = __builtin_amdgcn_wmma_f32_16x16x32_f16(
          false, a1, false, bf, (short)0, acc[j][1], false, false);
    }
  }

  int oOff = (lane & 16) ? 8 : 0;                    // C/D f32 layout
#pragma unroll
  for (int j = 0; j < 3; ++j) {
    int h = ty*TH + rowj[j];
    int w = tx*TW + cgj[j]*16 + laneN;
#pragma unroll
    for (int m = 0; m < 2; ++m)
#pragma unroll
      for (int r = 0; r < 8; ++r) {
        int o = m*16 + r + oOff;
        out[((size_t)(b*COUT + o)*H_ + h)*W_ + w] =
            acc[j][m][r] * (1.0f/WSCALE) + Cc[b*COUT + o];
      }
  }
}

// ---------------------------------------------------------------------------
extern "C" void kernel_launch(void* const* d_in, const int* in_sizes, int n_in,
                              void* d_out, int out_size, void* d_ws, size_t ws_size,
                              hipStream_t stream) {
  const float* x     = (const float*)d_in[0];
  const float* w1    = (const float*)d_in[1];
  const float* b1    = (const float*)d_in[2];
  const float* w2    = (const float*)d_in[3];
  const float* b2    = (const float*)d_in[4];
  const float* w3    = (const float*)d_in[5];
  const float* b3    = (const float*)d_in[6];
  const float* w4    = (const float*)d_in[7];
  const float* b4    = (const float*)d_in[8];
  const float* w5    = (const float*)d_in[9];
  const float* b5    = (const float*)d_in[10];
  const float* gcnw  = (const float*)d_in[11];
  const float* gcnb  = (const float*)d_in[12];
  const float* attn  = (const float*)d_in[13];
  const float* fw    = (const float*)d_in[14];
  const float* fb    = (const float*)d_in[15];
  float* out = (float*)d_out;

  char* ws = (char*)d_ws;
  float*    wsStats = (float*)(ws + WS_STATS);
  float*    wsS     = (float*)(ws + WS_S);
  float*    wsWm    = (float*)(ws + WS_WM);
  float*    wsG     = (float*)(ws + WS_G);
  float*    wsC     = (float*)(ws + WS_C);
  float*    wsFg    = (float*)(ws + WS_FG);
  _Float16* wsA     = (_Float16*)(ws + WS_WBA);

  k1_stats<<<256, 256, 0, stream>>>(x, wsStats);
  k2a_shift<<<1, 256, 0, stream>>>(x, wsStats, wsS);
  k2b_fold<<<1, 256, 0, stream>>>(wsS, w1, b1, w2, b2, w3, b3, w4, b4, w5, b5,
                                  gcnw, gcnb, attn, fw, fb,
                                  wsWm, wsG, wsC, wsFg);
  k2c_frag<<<800, 256, 0, stream>>>(wsFg, wsWm, wsA);
  k3_conv<<<dim3(W_/TW, H_/TH, B_), 256, 0, stream>>>(x, wsA, wsC, out);
}
